// PointModel_18717467476194
// MI455X (gfx1250) — matrix operand order repaired and checked by reading, not compiled, not run on hardware
//
#include <hip/hip_runtime.h>

typedef __attribute__((ext_vector_type(16))) _Float16 v16h;
typedef __attribute__((ext_vector_type(8)))  float    v8f;

#define NB   32
#define SEQ  1024
#define DIM  256
#define KDIM 64
#define NH   4
#define KHD  16
#define DVD  64
#define NL   4
#define BSR  (NB*SEQ)          // 32768 rows
#define EPSV 1e-5f
#define SCALEV 0.125f          // 1/sqrt(64)

// ---------------- WMMA fragment loaders (wave32, 16x16x32 f16) ----------------
// A (16xK row-major): lane m=lane&15 holds K = c0 + {0..7, 16..23}, c0 = 8*(lane>=16)
__device__ __forceinline__ v16h load_a_frag(const _Float16* p, int stride, int lane) {
  int m  = lane & 15;
  int c0 = (lane >> 4) << 3;
  const _Float16* q = p + m * stride + c0;
  union { uint4 u[2]; v16h v; } r;
  r.u[0] = *(const uint4*)(q);
  r.u[1] = *(const uint4*)(q + 16);
  return r.v;
}
// B (KxN), sourced from B^T row-major [N x K]: lane n=lane&15 holds K = k0 + {0..7, 16..23},
// k0 = 8*(lane>=16)  (per ISA B-matrix layout tables)
__device__ __forceinline__ v16h load_b_frag(const _Float16* p, int stride, int lane) {
  int n  = lane & 15;
  int k0 = (lane >> 4) << 3;
  const _Float16* q = p + n * stride + k0;
  union { uint4 u[2]; v16h v; } r;
  r.u[0] = *(const uint4*)(q);
  r.u[1] = *(const uint4*)(q + 16);
  return r.v;
}
__device__ __forceinline__ v8f wmma16(v16h a, v16h b, v8f c) {
  return __builtin_amdgcn_wmma_f32_16x16x32_f16(false, a, false, b, (short)0, c, false, false);
}

// ---------------- generic WMMA GEMM: C = act((A @ Bt^T) + bias) * scale ----------------
// A: [M,Kd] f16 row-major; Bt: [N,Kd] f16 row-major (i.e. W^T).
// mode 0: f32 row-major [M,N]; mode 1: f16 row-major; mode 2: f16 transposed per batch
//         (out16[(b*256+n)*1024 + s], m = b*1024+s, requires N==256).
__global__ __launch_bounds__(128) void gemm_wmma(
    const _Float16* __restrict__ A, const _Float16* __restrict__ Bt,
    const float* __restrict__ bias, float* __restrict__ out32,
    _Float16* __restrict__ out16, int M, int N, int Kd,
    float scale, int relu, int mode)
{
  int wid  = blockIdx.x * (blockDim.x >> 5) + (threadIdx.x >> 5);
  int lane = threadIdx.x & 31;
  int tiles_n = N >> 4;
  int tiles   = (M >> 4) * tiles_n;
  if (wid >= tiles) return;
  int m0 = (wid / tiles_n) << 4;
  int n0 = (wid % tiles_n) << 4;
  const _Float16* Ab = A  + (size_t)m0 * Kd;
  const _Float16* Bb = Bt + (size_t)n0 * Kd;
  v8f c = {};
  for (int kk = 0; kk < Kd; kk += 32) {
    v16h af = load_a_frag(Ab + kk, Kd, lane);
    v16h bf = load_b_frag(Bb + kk, Kd, lane);
    c = wmma16(af, bf, c);
  }
  int col    = lane & 15;
  int rowoff = (lane >> 4) << 3;
  float bv = bias ? bias[n0 + col] : 0.0f;
#pragma unroll
  for (int r = 0; r < 8; ++r) {
    float v = (c[r] + bv) * scale;
    if (relu) v = v > 0.0f ? v : 0.0f;
    int gm = m0 + r + rowoff;
    if (mode == 0)      out32[(size_t)gm * N + n0 + col] = v;
    else if (mode == 1) out16[(size_t)gm * N + n0 + col] = (_Float16)v;
    else {
      int b = gm >> 10, s = gm & 1023;
      out16[(((size_t)b << 8) + (size_t)(n0 + col)) * 1024 + s] = (_Float16)v;
    }
  }
}

// ---------------- fused sigmoid-attention: attn = sigmoid(k q^T + bias) @ v ----------------
// block = 4 waves, handles 16 rows (i-tile) of one batch; wave w owns d-cols [64w,64w+64)
__global__ __launch_bounds__(128) void attn_kernel(
    const _Float16* __restrict__ kh, const _Float16* __restrict__ qh,
    const _Float16* __restrict__ vT, const float* __restrict__ biasb,
    float* __restrict__ attn)
{
  __shared__ _Float16 sS[16 * 64];   // 16 rows x 64 j, row-major
  int blk = blockIdx.x;
  int b   = blk >> 6;                // SEQ/16 = 64 tiles per batch
  int i0  = (blk & 63) << 4;
  int lane = threadIdx.x & 31;
  int w    = threadIdx.x >> 5;
  int m    = lane & 15;
  int rowoff = (lane >> 4) << 3;

  const _Float16* kbase = kh + (size_t)(b * SEQ + i0) * KDIM;
  v16h a0 = load_a_frag(kbase,      KDIM, lane);   // K 0..31
  v16h a1 = load_a_frag(kbase + 32, KDIM, lane);   // K 32..63

  float bias_r[8];
#pragma unroll
  for (int r = 0; r < 8; ++r) bias_r[r] = biasb[b * SEQ + i0 + r + rowoff];

  v8f acc[4] = {v8f{}, v8f{}, v8f{}, v8f{}};
  int d0 = w << 6;
  const _Float16* vbase = vT + ((size_t)b * DIM + d0) * SEQ;

  for (int j0 = 0; j0 < SEQ; j0 += 64) {
    // scores for this wave's 16-wide j sub-block
    const _Float16* qb = qh + (size_t)(b * SEQ + j0 + (w << 4)) * KDIM;
    v16h b0 = load_b_frag(qb,      KDIM, lane);
    v16h b1 = load_b_frag(qb + 32, KDIM, lane);
    v8f c = {};
    c = wmma16(a0, b0, c);
    c = wmma16(a1, b1, c);
#pragma unroll
    for (int r = 0; r < 8; ++r) {
      float t = c[r] + bias_r[r];
      t = 1.0f / (1.0f + __expf(-t));
      sS[(r + rowoff) * 64 + (w << 4) + m] = (_Float16)t;
    }
    __syncthreads();
    // attn accumulation: A = scores[16x64] from LDS, B = vT rows (contiguous)
#pragma unroll
    for (int jc = 0; jc < 2; ++jc) {
      v16h sa = load_a_frag(sS + jc * 32, 64, lane);
#pragma unroll
      for (int nt = 0; nt < 4; ++nt) {
        v16h vb = load_b_frag(vbase + (size_t)(nt << 4) * SEQ + j0 + jc * 32, SEQ, lane);
        acc[nt] = wmma16(sa, vb, acc[nt]);
      }
    }
    __syncthreads();
  }
  float* obase = attn + (size_t)(b * SEQ + i0) * DIM + d0;
#pragma unroll
  for (int nt = 0; nt < 4; ++nt)
#pragma unroll
    for (int r = 0; r < 8; ++r)
      obase[(size_t)(r + rowoff) * DIM + (nt << 4) + m] = acc[nt][r];
}

// ---------------- add + LayerNorm (optionally affine); writes f32 + f16 ----------------
__global__ __launch_bounds__(256) void add_ln(
    const float* __restrict__ A, const float* __restrict__ Bv,
    const float* __restrict__ gamma, const float* __restrict__ beta,
    float* __restrict__ out32, _Float16* __restrict__ out16)
{
  __shared__ float red1[8], red2[8];
  size_t row = blockIdx.x;
  int tid = threadIdx.x;
  float v = A[row * DIM + tid] + Bv[row * DIM + tid];
  float s = v;
  for (int off = 16; off > 0; off >>= 1) s += __shfl_down(s, off, 32);
  if ((tid & 31) == 0) red1[tid >> 5] = s;
  __syncthreads();
  float tot = 0.f;
#pragma unroll
  for (int i = 0; i < 8; ++i) tot += red1[i];
  float mean = tot * (1.0f / DIM);
  float d  = v - mean;
  float sq = d * d;
  for (int off = 16; off > 0; off >>= 1) sq += __shfl_down(sq, off, 32);
  if ((tid & 31) == 0) red2[tid >> 5] = sq;
  __syncthreads();
  float vt = 0.f;
#pragma unroll
  for (int i = 0; i < 8; ++i) vt += red2[i];
  float y = d * rsqrtf(vt * (1.0f / DIM) + EPSV);
  if (gamma) y = y * gamma[tid] + beta[tid];
  out32[row * DIM + tid] = y;
  out16[row * DIM + tid] = (_Float16)y;
}

// ---------------- bias GEMV: bias[b,s] = dot(x_row, Wb) + bb ----------------
__global__ __launch_bounds__(256) void bias_gemv(
    const float* __restrict__ x, const float* __restrict__ Wb,
    const float* __restrict__ bb, float* __restrict__ outb)
{
  int row  = blockIdx.x * 8 + (threadIdx.x >> 5);
  int lane = threadIdx.x & 31;
  const float* xr = x + (size_t)row * DIM;
  float s = 0.f;
#pragma unroll
  for (int j = 0; j < 8; ++j) s += xr[lane + j * 32] * Wb[lane + j * 32];
  for (int off = 16; off > 0; off >>= 1) s += __shfl_down(s, off, 32);
  if (lane == 0) outb[row] = s + bb[0];
}

// ---------------- pooling: scores + softmax over S ----------------
__global__ __launch_bounds__(256) void pool_softmax(
    const float* __restrict__ pk, const float* __restrict__ queries,
    float* __restrict__ scb)
{
  __shared__ float redm[8], reds[8];
  int b = blockIdx.x >> 2, h = blockIdx.x & 3;
  int tid = threadIdx.x;
  float qv[KHD];
#pragma unroll
  for (int k = 0; k < KHD; ++k) qv[k] = queries[h * KHD + k];
  float sc[4];
#pragma unroll
  for (int t = 0; t < 4; ++t) {
    const float* p = pk + ((size_t)b * SEQ + tid + t * 256) * KHD;
    float a = 0.f;
#pragma unroll
    for (int k = 0; k < KHD; ++k) a += p[k] * qv[k];
    sc[t] = a;
  }
  float mx = fmaxf(fmaxf(sc[0], sc[1]), fmaxf(sc[2], sc[3]));
  for (int off = 16; off > 0; off >>= 1) mx = fmaxf(mx, __shfl_xor(mx, off, 32));
  if ((tid & 31) == 0) redm[tid >> 5] = mx;
  __syncthreads();
  float M = -1e30f;
#pragma unroll
  for (int i = 0; i < 8; ++i) M = fmaxf(M, redm[i]);
  float ls = 0.f;
#pragma unroll
  for (int t = 0; t < 4; ++t) { sc[t] = __expf(sc[t] - M); ls += sc[t]; }
  for (int off = 16; off > 0; off >>= 1) ls += __shfl_xor(ls, off, 32);
  if ((tid & 31) == 0) reds[tid >> 5] = ls;
  __syncthreads();
  float S = 0.f;
#pragma unroll
  for (int i = 0; i < 8; ++i) S += reds[i];
  float inv = 1.0f / S;
#pragma unroll
  for (int t = 0; t < 4; ++t)
    scb[((size_t)b * NH + h) * SEQ + tid + t * 256] = sc[t] * inv;
}

__global__ __launch_bounds__(64) void pool_reduce(
    const float* __restrict__ scb, const float* __restrict__ pv,
    float* __restrict__ pooled)
{
  int b = blockIdx.x >> 2, h = blockIdx.x & 3;
  int d = threadIdx.x;
  const float* sp = scb + ((size_t)b * NH + h) * SEQ;
  float s = 0.f;
  for (int si = 0; si < SEQ; ++si)
    s += sp[si] * pv[((size_t)b * SEQ + si) * DVD + d];
  pooled[b * DIM + h * DVD + d] = s;
}

__global__ __launch_bounds__(64) void final_lin(
    const float* __restrict__ pooled, const float* __restrict__ W,
    const float* __restrict__ bb, float* __restrict__ out)
{
  int b = threadIdx.x >> 1, j = threadIdx.x & 1;
  float s = bb[j];
  for (int d = 0; d < DIM; ++d) s += pooled[b * DIM + d] * W[d * 2 + j];
  out[b * 2 + j] = s;
}

// ---------------- prep kernels ----------------
__global__ void transpose_w(const float* __restrict__ W, _Float16* __restrict__ Wt,
                            int Drows, int Ncols) {
  int i = blockIdx.x * 256 + threadIdx.x;
  if (i >= Drows * Ncols) return;
  int n = i / Drows, d = i - n * Drows;
  Wt[(size_t)n * Drows + d] = (_Float16)W[(size_t)d * Ncols + n];
}
__global__ void cvt_copy(const float* __restrict__ in, float* __restrict__ x32,
                         _Float16* __restrict__ x16, int n) {
  int i = blockIdx.x * 256 + threadIdx.x;
  if (i >= n) return;
  float v = in[i];
  x32[i] = v;
  x16[i] = (_Float16)v;
}

// ---------------- host launcher ----------------
extern "C" void kernel_launch(void* const* d_in, const int* in_sizes, int n_in,
                              void* d_out, int out_size, void* d_ws, size_t ws_size,
                              hipStream_t stream) {
  const float* input   = (const float*)d_in[0];
  const float* enc_Wk  = (const float*)d_in[3];
  const float* enc_bk  = (const float*)d_in[4];
  const float* enc_Wq  = (const float*)d_in[5];
  const float* enc_bq  = (const float*)d_in[6];
  const float* enc_Wv  = (const float*)d_in[7];
  const float* enc_bv  = (const float*)d_in[8];
  const float* enc_Wb  = (const float*)d_in[9];
  const float* enc_bb  = (const float*)d_in[10];
  const float* enc_Wff = (const float*)d_in[11];
  const float* enc_bff = (const float*)d_in[12];
  const float* enc_g1  = (const float*)d_in[13];
  const float* enc_be1 = (const float*)d_in[14];
  const float* pool_Wk = (const float*)d_in[15];
  const float* pool_bk = (const float*)d_in[16];
  const float* pool_Wv = (const float*)d_in[17];
  const float* pool_bv = (const float*)d_in[18];
  const float* queries = (const float*)d_in[19];
  const float* lin1_W  = (const float*)d_in[20];
  const float* lin1_b  = (const float*)d_in[21];
  float* out = (float*)d_out;

  char* ws = (char*)d_ws;
  const size_t SZ_X = (size_t)BSR * DIM * 4, SZ_XH = (size_t)BSR * DIM * 2;
  const size_t SZ_KH = (size_t)BSR * KDIM * 2;
  size_t o_x = 0, o_xh = o_x + SZ_X, o_kh = o_xh + SZ_XH, o_qh = o_kh + SZ_KH;
  size_t o_vT = o_qh + SZ_KH, o_bias = o_vT + SZ_XH, o_attn = o_bias + (size_t)BSR * 4;
  size_t o_res = o_attn + SZ_X, o_resh = o_res + SZ_X, o_w = o_resh + SZ_XH;
  const size_t WL = 327680;                 // per-layer transposed weights (bytes)
  size_t o_pwk = o_w + NL * WL, o_pwv = o_pwk + 8192;
  size_t o_pk = o_kh, o_pv = o_qh;          // pooling aliases (k/q dead by then)
  size_t o_ff = o_attn, o_scb = o_attn, o_pooled = o_attn + 524288;

  float*    x32  = (float*)(ws + o_x);
  _Float16* xh   = (_Float16*)(ws + o_xh);
  _Float16* khb  = (_Float16*)(ws + o_kh);
  _Float16* qhb  = (_Float16*)(ws + o_qh);
  _Float16* vT   = (_Float16*)(ws + o_vT);
  float*    bsb  = (float*)(ws + o_bias);
  float*    attn = (float*)(ws + o_attn);
  float*    res  = (float*)(ws + o_res);
  _Float16* resh = (_Float16*)(ws + o_resh);
  float*    ff   = (float*)(ws + o_ff);
  float*    pkb  = (float*)(ws + o_pk);
  float*    pvb  = (float*)(ws + o_pv);
  float*    scb  = (float*)(ws + o_scb);
  float*    pooled = (float*)(ws + o_pooled);

  auto gemm = [&](const _Float16* A, const _Float16* Bt, const float* bias,
                  float* o32, _Float16* o16, int M, int N, float scale,
                  int relu, int mode) {
    int tiles = (M / 16) * (N / 16);
    gemm_wmma<<<(tiles + 3) / 4, 128, 0, stream>>>(A, Bt, bias, o32, o16, M, N,
                                                   DIM, scale, relu, mode);
  };
  auto xpose = [&](const float* W, _Float16* Wt, int N) {
    int n = DIM * N;
    transpose_w<<<(n + 255) / 256, 256, 0, stream>>>(W, Wt, DIM, N);
  };

  // init x / xh
  cvt_copy<<<(BSR * DIM + 255) / 256, 256, 0, stream>>>(input, x32, xh, BSR * DIM);

  // transpose + convert weights
  for (int i = 0; i < NL; ++i) {
    char* wl = ws + o_w + (size_t)i * WL;
    xpose(enc_Wk  + (size_t)i * DIM * KDIM, (_Float16*)(wl),           KDIM);
    xpose(enc_Wq  + (size_t)i * DIM * KDIM, (_Float16*)(wl + 32768),   KDIM);
    xpose(enc_Wv  + (size_t)i * DIM * DIM,  (_Float16*)(wl + 65536),   DIM);
    xpose(enc_Wff + (size_t)i * DIM * DIM,  (_Float16*)(wl + 196608),  DIM);
  }
  xpose(pool_Wk, (_Float16*)(ws + o_pwk), KHD);
  xpose(pool_Wv, (_Float16*)(ws + o_pwv), DVD);

  for (int i = 0; i < NL; ++i) {
    char* wl = ws + o_w + (size_t)i * WL;
    const _Float16* Wkt  = (const _Float16*)(wl);
    const _Float16* Wqt  = (const _Float16*)(wl + 32768);
    const _Float16* Wvt  = (const _Float16*)(wl + 65536);
    const _Float16* Wfft = (const _Float16*)(wl + 196608);

    gemm(xh, Wkt, enc_bk + i * KDIM, nullptr, khb, BSR, KDIM, SCALEV, 0, 1);
    gemm(xh, Wqt, enc_bq + i * KDIM, nullptr, qhb, BSR, KDIM, 1.0f,  0, 1);
    gemm(xh, Wvt, enc_bv + i * DIM,  nullptr, vT,  BSR, DIM,  1.0f,  0, 2);
    bias_gemv<<<BSR / 8, 256, 0, stream>>>(x32, enc_Wb + (size_t)i * DIM,
                                           enc_bb + i, bsb);
    attn_kernel<<<NB * (SEQ / 16), 128, 0, stream>>>(khb, qhb, vT, bsb, attn);
    add_ln<<<BSR, 256, 0, stream>>>(x32, attn, enc_g1 + i * DIM,
                                    enc_be1 + i * DIM, res, resh);
    gemm(resh, Wfft, enc_bff + i * DIM, ff, nullptr, BSR, DIM, 1.0f, 1, 0);
    add_ln<<<BSR, 256, 0, stream>>>(res, ff, nullptr, nullptr, x32, xh);
  }

  // pooling head
  gemm(xh, (const _Float16*)(ws + o_pwk), pool_bk, pkb, nullptr, BSR, KHD, SCALEV, 0, 0);
  gemm(xh, (const _Float16*)(ws + o_pwv), pool_bv, pvb, nullptr, BSR, DVD, 1.0f,  0, 0);
  pool_softmax<<<NB * NH, 256, 0, stream>>>(pkb, queries, scb);
  pool_reduce<<<NB * NH, 64, 0, stream>>>(scb, pvb, pooled);
  final_lin<<<1, 64, 0, stream>>>(pooled, lin1_W, lin1_b, out);
}
